// MultiHeadAttentionBlock_5068061409795
// MI455X (gfx1250) — compile-verified, matrix operand order checked by reading, not run on hardware
//
#include <hip/hip_runtime.h>

// ---------------------------------------------------------------------------
// MI455X (gfx1250, wave32) multi-head attention block.
// All matmuls on v_wmma_f32_16x16x32_bf16 (fp32 accumulate), softmax fp32.
// LDS staging via GLOBAL_LOAD_ASYNC_TO_LDS (ASYNCcnt) when the toolchain
// exposes the builtins; synchronous global->VGPR->ds fallback otherwise.
// ---------------------------------------------------------------------------

typedef __bf16 bf16_t;
typedef __attribute__((ext_vector_type(16))) __bf16 v16bf;
typedef __attribute__((ext_vector_type(8)))  __bf16 v8bf;
typedef __attribute__((ext_vector_type(8)))  float   v8f;
typedef __attribute__((ext_vector_type(4)))  float   v4f;
typedef __attribute__((ext_vector_type(4)))  int     v4i;

#define DEV __device__ __forceinline__

static constexpr int Bb = 4;      // batch
static constexpr int Ss = 2048;   // sequence
static constexpr int Dm = 1024;   // d_model
static constexpr int Hh = 16;     // heads
static constexpr int Dk = 64;     // head dim

#if defined(__has_builtin)
#if __has_builtin(__builtin_amdgcn_global_load_async_to_lds_b128) && \
    __has_builtin(__builtin_amdgcn_s_wait_asynccnt)
#define HAVE_ASYNC_LDS 1
#endif
#endif

#ifdef HAVE_ASYNC_LDS
// prototype (from hipcc diagnostic): first arg is  v4i __attribute__((address_space(1)))*
typedef v4i __attribute__((address_space(1)))* gv4i_p;  // global
typedef v4i __attribute__((address_space(3)))* lv4i_p;  // LDS
DEV void async_cp16(const bf16_t* g, bf16_t* l) {
  __builtin_amdgcn_global_load_async_to_lds_b128((gv4i_p)(g), (lv4i_p)(l), 0, 0);
}
DEV void async_wait0() { __builtin_amdgcn_s_wait_asynccnt(0); }
#endif

DEV v8f wmma_bf16(v16bf a, v16bf b, v8f c) {
  // (neg_a, A, neg_b, B, c_mod, C, reuse_a, reuse_b)
  return __builtin_amdgcn_wmma_f32_16x16x32_bf16(false, a, false, b, (short)0, c,
                                                 false, false);
}

DEV v16bf make16(v8bf lo, v8bf hi) {
  return __builtin_shufflevector(lo, hi, 0, 1, 2, 3, 4, 5, 6, 7, 8, 9, 10, 11,
                                 12, 13, 14, 15);
}

// 16x32 bf16 A/B fragment from a row-major tile t[rows][ldk].
// ISA 16-bit A layout: lanes 0-15 hold row r0+L, K chunks {k0+0..7, k0+16..23};
// lanes 16-31 hold row r0+L-16, K chunks {k0+8..15, k0+24..31}.
DEV v16bf frag_ld(const bf16_t* t, int r0, int ldk, int k0, int lane) {
  const bf16_t* p = t + (size_t)(r0 + (lane & 15)) * ldk + k0 + (lane >> 4) * 8;
  return make16(*(const v8bf*)p, *(const v8bf*)(p + 16));
}

// ---------------------------------------------------------------------------
// fp32 -> bf16 conversion (weights, converted once; re-read 64x by GEMM tiles)
// ---------------------------------------------------------------------------
__global__ void cvt_f32_bf16(const float* __restrict__ x, bf16_t* __restrict__ y,
                             int n) {
  int i = blockIdx.x * blockDim.x + threadIdx.x;
  if (i < n) y[i] = (bf16_t)x[i];
}

// ---------------------------------------------------------------------------
// GEMM: out[M,N] = A[M,K] @ W[N,K]^T + bias[N]   (nn.Linear)
// Block tile 128x128x64, 256 threads = 8 waves; wave computes 64x32.
// AT = float (converted to bf16 while staging to LDS) or bf16_t (async copy).
// OT = bf16_t (intermediate activations) or float (final output).
// ---------------------------------------------------------------------------
template <typename AT, typename OT>
__global__ __launch_bounds__(256) void gemm_bias_wmma(
    const AT* __restrict__ A, const bf16_t* __restrict__ W,
    const float* __restrict__ bias, OT* __restrict__ out, int M, int N, int K) {
  __shared__ bf16_t As[128 * 64];
  __shared__ bf16_t Wsh[128 * 64];

  const int tid = threadIdx.x, lane = tid & 31, wid = tid >> 5;
  const int wm = (wid >> 2) * 64;  // wave row origin in block tile
  const int wn = (wid & 3) * 32;   // wave col origin in block tile
  const int mb = blockIdx.y * 128, nb = blockIdx.x * 128;

  v8f acc[4][2] = {};

  for (int kt = 0; kt < K; kt += 64) {
    // stage: 128x64 bf16 each for A and W; 1024 8-elem chunks, 4 per thread
    for (int i = 0; i < 4; ++i) {
      int c = tid + 256 * i;
      int row = c >> 3, col = (c & 7) * 8;
      if constexpr (sizeof(AT) == 4) {
        const float* ap = (const float*)A + (size_t)(mb + row) * K + kt + col;
        v4f f0 = *(const v4f*)ap;
        v4f f1 = *(const v4f*)(ap + 4);
        v8bf hv;
        for (int e = 0; e < 4; ++e) {
          hv[e] = (bf16_t)f0[e];
          hv[e + 4] = (bf16_t)f1[e];
        }
        *(v8bf*)(As + row * 64 + col) = hv;
        if (kt + 64 < K) __builtin_prefetch(ap + 64, 0, 0);
      } else {
        const bf16_t* ap = (const bf16_t*)A + (size_t)(mb + row) * K + kt + col;
#ifdef HAVE_ASYNC_LDS
        async_cp16(ap, As + row * 64 + col);
#else
        *(v8bf*)(As + row * 64 + col) = *(const v8bf*)ap;
#endif
        if (kt + 64 < K) __builtin_prefetch(ap + 64, 0, 0);
      }
      const bf16_t* wp = W + (size_t)(nb + row) * K + kt + col;
#ifdef HAVE_ASYNC_LDS
      async_cp16(wp, Wsh + row * 64 + col);
#else
      *(v8bf*)(Wsh + row * 64 + col) = *(const v8bf*)wp;
#endif
      if (kt + 64 < K) __builtin_prefetch(wp + 64, 0, 0);
    }
#ifdef HAVE_ASYNC_LDS
    async_wait0();
#endif
    __syncthreads();

    for (int ks = 0; ks < 2; ++ks) {
      v16bf af[4], wf[2];
      for (int i = 0; i < 4; ++i) af[i] = frag_ld(As, wm + i * 16, 64, ks * 32, lane);
      for (int j = 0; j < 2; ++j) wf[j] = frag_ld(Wsh, wn + j * 16, 64, ks * 32, lane);
      for (int i = 0; i < 4; ++i)
        for (int j = 0; j < 2; ++j) acc[i][j] = wmma_bf16(af[i], wf[j], acc[i][j]);
    }
    __syncthreads();
  }

  // epilogue: C fragment layout -> row = base + e + 8*(lane/16), col = lane%16
  const int hlf = lane >> 4, l15 = lane & 15;
  for (int j = 0; j < 2; ++j) {
    int col = nb + wn + j * 16 + l15;
    float bv = bias[col];
    for (int i = 0; i < 4; ++i)
      for (int e = 0; e < 8; ++e) {
        int row = mb + wm + i * 16 + e + 8 * hlf;
        float vv = acc[i][j][e] + bv;
        out[(size_t)row * N + col] = (OT)vv;
      }
  }
}

// ---------------------------------------------------------------------------
// Flash attention over one (b, h, 64-row query block).
// 128 threads = 4 waves; wave w owns query rows [w*16, w*16+16).
// KV streamed in 64-wide tiles through LDS; online softmax fp32.
// K tile staged with async-to-LDS copies (byte-exact); V needs a transpose so
// it bounces through VGPRs into a [dk][kv] LDS layout for P@V B-fragments.
// ---------------------------------------------------------------------------
__global__ __launch_bounds__(128) void mha_flash(const bf16_t* __restrict__ qp,
                                                 const bf16_t* __restrict__ kp,
                                                 const bf16_t* __restrict__ vp,
                                                 bf16_t* __restrict__ ctx) {
  __shared__ bf16_t kt[64 * 64];     // K tile, row-major [kv][dk]
  __shared__ bf16_t vt[64 * 64];     // V tile transposed  [dk->n][kv->k]
  __shared__ bf16_t pt[4][16 * 64];  // per-wave P relayout bounce

  const int tid = threadIdx.x, lane = tid & 31, wid = tid >> 5;
  const int hlf = lane >> 4, l15 = lane & 15;
  const int b = blockIdx.y >> 4, h = blockIdx.y & 15;
  const int q0 = blockIdx.x * 64;
  const size_t hd = (size_t)h * Dk;

  // Q fragments (16 rows x 64 dk = 2 A-frags), loaded straight from global
  const bf16_t* qb = qp + ((size_t)b * Ss + q0 + wid * 16) * Dm + hd;
  v16bf qa[2];
  for (int ks = 0; ks < 2; ++ks) {
    const bf16_t* p = qb + (size_t)l15 * Dm + ks * 32 + hlf * 8;
    qa[ks] = make16(*(const v8bf*)p, *(const v8bf*)(p + 16));
  }

  v8f o[4] = {};
  float m_r[8], l_r[8];
  for (int j = 0; j < 8; ++j) {
    m_r[j] = -3.0e38f;
    l_r[j] = 0.f;
  }

  for (int t = 0; t < Ss / 64; ++t) {
    // stage K row-major (async), V transposed (VGPR bounce)
    for (int i = 0; i < 4; ++i) {
      int c = tid + 128 * i;  // 512 chunks of 8 bf16
      int row = c >> 3, col = (c & 7) * 8;
      const size_t grow = ((size_t)b * Ss + t * 64 + row) * Dm + hd + col;
#ifdef HAVE_ASYNC_LDS
      async_cp16(kp + grow, kt + row * 64 + col);
#else
      *(v8bf*)(kt + row * 64 + col) = *(const v8bf*)(kp + grow);
#endif
      v8bf vv = *(const v8bf*)(vp + grow);
      for (int e = 0; e < 8; ++e) vt[(col + e) * 64 + row] = vv[e];
      if (t + 1 < Ss / 64) {
        __builtin_prefetch(kp + grow + (size_t)64 * Dm, 0, 0);
        __builtin_prefetch(vp + grow + (size_t)64 * Dm, 0, 0);
      }
    }
#ifdef HAVE_ASYNC_LDS
    async_wait0();
#endif
    __syncthreads();

    // S = (Q K^T) * 1/sqrt(dk)
    v8f s[4] = {};
    for (int ks = 0; ks < 2; ++ks)
      for (int f = 0; f < 4; ++f)
        s[f] = wmma_bf16(qa[ks], frag_ld(kt, f * 16, 64, ks * 32, lane), s[f]);
    for (int f = 0; f < 4; ++f)
      for (int j = 0; j < 8; ++j) s[f][j] *= 0.125f;

    // online softmax: row j+8*hlf lives in VGPR j across a 16-lane N-group
    float alpha[8];
    for (int j = 0; j < 8; ++j) {
      float mx = fmaxf(fmaxf(s[0][j], s[1][j]), fmaxf(s[2][j], s[3][j]));
      for (int d = 1; d < 16; d <<= 1) mx = fmaxf(mx, __shfl_xor(mx, d, 32));
      float mnew = fmaxf(m_r[j], mx);
      alpha[j] = __expf(m_r[j] - mnew);
      m_r[j] = mnew;
      float rs = 0.f;
      for (int f = 0; f < 4; ++f) {
        float pe = __expf(s[f][j] - mnew);
        s[f][j] = pe;
        rs += pe;
      }
      for (int d = 1; d < 16; d <<= 1) rs += __shfl_xor(rs, d, 32);
      l_r[j] = l_r[j] * alpha[j] + rs;
      for (int f = 0; f < 4; ++f) o[f][j] *= alpha[j];
    }

    // P (C-frag layout) -> per-wave LDS -> A-frag layout
    bf16_t* pw = pt[wid];
    for (int f = 0; f < 4; ++f)
      for (int j = 0; j < 8; ++j)
        pw[(j + 8 * hlf) * 64 + f * 16 + l15] = (bf16_t)s[f][j];
    asm volatile("s_wait_dscnt 0x0" ::: "memory");  // same-wave LDS RAW

    v16bf pa[2];
    for (int ks = 0; ks < 2; ++ks) pa[ks] = frag_ld(pw, 0, 64, ks * 32, lane);

    // O += P @ V
    for (int ks = 0; ks < 2; ++ks)
      for (int f = 0; f < 4; ++f)
        o[f] = wmma_bf16(pa[ks], frag_ld(vt, f * 16, 64, ks * 32, lane), o[f]);
    __syncthreads();
  }

  // normalize and write ctx (bf16, [B,S,D] with head offset)
  for (int f = 0; f < 4; ++f)
    for (int j = 0; j < 8; ++j) {
      size_t row = (size_t)b * Ss + q0 + wid * 16 + j + 8 * hlf;
      ctx[row * Dm + hd + f * 16 + l15] = (bf16_t)(o[f][j] / l_r[j]);
    }
}

// ---------------------------------------------------------------------------
// Launch: cvt weights -> QKV projections -> flash attention -> out projection
// ---------------------------------------------------------------------------
extern "C" void kernel_launch(void* const* d_in, const int* in_sizes, int n_in,
                              void* d_out, int out_size, void* d_ws,
                              size_t ws_size, hipStream_t stream) {
  const float* q  = (const float*)d_in[0];
  const float* k  = (const float*)d_in[1];
  const float* v  = (const float*)d_in[2];
  const float* wq = (const float*)d_in[3];
  const float* bq = (const float*)d_in[4];
  const float* wk = (const float*)d_in[5];
  const float* bk = (const float*)d_in[6];
  const float* wv = (const float*)d_in[7];
  const float* bv = (const float*)d_in[8];
  const float* wo = (const float*)d_in[9];
  const float* bo = (const float*)d_in[10];
  float* out = (float*)d_out;

  // workspace (bf16 elements): 4 weights (1M each) + qp/kp/vp/ctx (8M each) = 72MB
  bf16_t* ws = (bf16_t*)d_ws;
  const size_t MW = (size_t)Dm * Dm;
  const size_t MA = (size_t)Bb * Ss * Dm;
  bf16_t* wqb = ws;
  bf16_t* wkb = ws + MW;
  bf16_t* wvb = ws + 2 * MW;
  bf16_t* wob = ws + 3 * MW;
  bf16_t* qpb = ws + 4 * MW;
  bf16_t* kpb = qpb + MA;
  bf16_t* vpb = kpb + MA;
  bf16_t* ctx = vpb + MA;

  const int nW = (int)MW;
  cvt_f32_bf16<<<(nW + 255) / 256, 256, 0, stream>>>(wq, wqb, nW);
  cvt_f32_bf16<<<(nW + 255) / 256, 256, 0, stream>>>(wk, wkb, nW);
  cvt_f32_bf16<<<(nW + 255) / 256, 256, 0, stream>>>(wv, wvb, nW);
  cvt_f32_bf16<<<(nW + 255) / 256, 256, 0, stream>>>(wo, wob, nW);

  const int M = Bb * Ss;  // 8192
  dim3 gg(Dm / 128, M / 128), gb(256);
  gemm_bias_wmma<float, bf16_t><<<gg, gb, 0, stream>>>(q, wqb, bq, qpb, M, Dm, Dm);
  gemm_bias_wmma<float, bf16_t><<<gg, gb, 0, stream>>>(k, wkb, bk, kpb, M, Dm, Dm);
  gemm_bias_wmma<float, bf16_t><<<gg, gb, 0, stream>>>(v, wvb, bv, vpb, M, Dm, Dm);

  mha_flash<<<dim3(Ss / 64, Bb * Hh), dim3(128), 0, stream>>>(qpb, kpb, vpb, ctx);

  gemm_bias_wmma<bf16_t, float><<<gg, gb, 0, stream>>>(ctx, wob, bo, out, M, Dm, Dm);
}
